// SineNet_2430951489779
// MI455X (gfx1250) — matrix operand order repaired and checked by reading, not compiled, not run on hardware
//
#include <hip/hip_runtime.h>
#include <hip/hip_bf16.h>
#include <stdint.h>

typedef __attribute__((ext_vector_type(16))) __bf16 v16bf;
typedef __attribute__((ext_vector_type(8)))  float  v8f;

#define BB 64
#define TT 1024
#define II 128
#define HH 512
#define NS 32
#define BT (BB*TT)      // 65536
#define G4 (4*HH)       // 2048

// ---------------- fragment helpers ----------------
// A-fragment (16x32 bf16, M x K) from row-major bf16 source.
// Per ISA layout: lane l -> m = l&15, K offset koff = (l>=16)?8:0;
// elements j=0..7 -> k = koff+j, j=8..15 -> k = koff+16+(j-8).
// => exactly two contiguous 16B runs per lane.
__device__ __forceinline__ v16bf load_afrag(const __bf16* base, int stride, int lane) {
    int m    = lane & 15;
    int koff = (lane >> 4) << 3;
    const __bf16* p = base + (size_t)m * stride + koff;
    v16bf a;
    ((uint4*)&a)[0] = *(const uint4*)(p);
    ((uint4*)&a)[1] = *(const uint4*)(p + 16);
    return a;
}

// B-fragment (32x16 bf16, K x N) from pre-packed buffer: frag f is 512
// contiguous bf16 in (lane, element) order -> one 32B vector load per lane.
__device__ __forceinline__ v16bf load_bfrag(const __bf16* packed, int f, int lane) {
    return *(const v16bf*)(packed + (((size_t)f << 5) + lane) * 16);
}

__device__ __forceinline__ float sigmoidf_(float x) {
    return 1.0f / (1.0f + __expf(-x));
}

// ---------------- weight packing: f32 row-major [K][N] -> bf16 B-frags ----
// dst linear index e: f = e/512, lane = (e/16)&31, j = e&15
// source: k = kc*32 + ((lane>=16)?8:0) + (j&7) + ((j>=8)?16:0), n = nt*16 + (lane&15)
__global__ void pack_b_kernel(const float* __restrict__ W, __bf16* __restrict__ dst,
                              int K, int N) {
    int total = K * N;
    int kchunks = K >> 5;
    for (int e = blockIdx.x * blockDim.x + threadIdx.x; e < total;
         e += gridDim.x * blockDim.x) {
        int f    = e >> 9;
        int lane = (e >> 4) & 31;
        int j    = e & 15;
        int nt = f / kchunks;
        int kc = f % kchunks;
        int k = (kc << 5) + ((lane >> 4) << 3) + (j & 7) + ((j >> 3) << 4);
        int n = (nt << 4) + (lane & 15);
        dst[e] = (__bf16)W[(size_t)k * N + n];
    }
}

__global__ void zero_f32_kernel(float* __restrict__ p, int n) {
    int i = blockIdx.x * blockDim.x + threadIdx.x;
    if (i < n) p[i] = 0.0f;
}

// ---------------- fused fc1 + gates_x GEMM ----------------
// One workgroup (4 waves) per 16-row M-tile of BT.
// Phase 1: fc1 tile = x[16x128] @ W1[128x512] + b1 -> LDS bf16 [16][512]
// Phase 2: gates_x tile = fc1 @ W_ih[512x2048] + b_lstm -> global bf16
__global__ void __launch_bounds__(128)
fused_fc1_gatesx(const float* __restrict__ x, const float* __restrict__ b1,
                 const float* __restrict__ b_lstm,
                 const __bf16* __restrict__ W1p, const __bf16* __restrict__ Wihp,
                 __bf16* __restrict__ gates_x) {
    __shared__ __align__(16) __bf16 fc1s[16 * HH];   // 16 KB

    const int tid  = threadIdx.x;
    const int lane = tid & 31;
    const int wave = tid >> 5;
    const int mrow0 = blockIdx.x * 16;
    const int n    = lane & 15;
    const int rsel = (lane >> 4) << 3;

    // ---- phase 1: build 4 A-frags from f32 x (convert to bf16) ----
    v16bf ax[4];
    {
        int m = lane & 15, koff = (lane >> 4) << 3;
        #pragma unroll
        for (int kc = 0; kc < 4; ++kc) {
            const float* p = x + (size_t)(mrow0 + m) * II + kc * 32 + koff;
            v16bf a;
            #pragma unroll
            for (int j = 0; j < 8; ++j) a[j] = (__bf16)p[j];
            #pragma unroll
            for (int j = 0; j < 8; ++j) a[8 + j] = (__bf16)p[16 + j];
            ax[kc] = a;
        }
    }
    #pragma unroll
    for (int i = 0; i < 8; ++i) {
        int nt = wave * 8 + i;                 // 32 N-tiles of 512
        v8f acc = {};
        #pragma unroll
        for (int kc = 0; kc < 4; ++kc) {
            v16bf b = load_bfrag(W1p, nt * 4 + kc, lane);
            acc = __builtin_amdgcn_wmma_f32_16x16x32_bf16(
                false, ax[kc], false, b, (short)0, acc, false, false);
        }
        float bias = b1[nt * 16 + n];
        #pragma unroll
        for (int r = 0; r < 8; ++r)
            fc1s[(r + rsel) * HH + nt * 16 + n] = (__bf16)(acc[r] + bias);
    }
    __syncthreads();

    // ---- phase 2: 2048 columns = 128 N-tiles, 32 per wave, K=512 ----
    for (int i = 0; i < 32; ++i) {
        int nt = wave * 32 + i;
        v8f acc = {};
        #pragma unroll
        for (int kc = 0; kc < 16; ++kc) {
            v16bf a = load_afrag(&fc1s[kc * 32], HH, lane);   // ds_load_b128 x2
            v16bf b = load_bfrag(Wihp, nt * 16 + kc, lane);
            acc = __builtin_amdgcn_wmma_f32_16x16x32_bf16(
                false, a, false, b, (short)0, acc, false, false);
        }
        float bias = b_lstm[nt * 16 + n];
        #pragma unroll
        for (int r = 0; r < 8; ++r)
            gates_x[(size_t)(mrow0 + r + rsel) * G4 + nt * 16 + n] =
                (__bf16)(acc[r] + bias);
    }
}

// ---------------- one LSTM timestep ----------------
// grid = 128 single-wave workgroups: block (mt, jt), mt in [0,4), jt in [0,32).
// Wave computes i/f/g/o 16x16 tiles for rows mt*16.. and h-cols jt*16..,
// so the cell update is fully register-resident.
__global__ void __launch_bounds__(32)
lstm_step(const __bf16* __restrict__ gates_x, const __bf16* __restrict__ Whhp,
          __bf16* __restrict__ h_all, float* __restrict__ c_state, int t) {
    const int lane = threadIdx.x & 31;
    const int mt = blockIdx.x & 3;
    const int jt = blockIdx.x >> 2;
    const int n    = lane & 15;
    const int rsel = (lane >> 4) << 3;

    v8f acc[4] = { {}, {}, {}, {} };
    if (t > 0) {
        // A rows are batch entries: row b lives at h_all[(b*TT + t-1)*HH]
        const __bf16* hbase = h_all + ((size_t)(mt * 16) * TT + (t - 1)) * HH;
        #pragma unroll 4
        for (int kc = 0; kc < 16; ++kc) {
            v16bf a = load_afrag(hbase + kc * 32, TT * HH, lane);
            #pragma unroll
            for (int g = 0; g < 4; ++g) {
                v16bf b = load_bfrag(Whhp, (g * 32 + jt) * 16 + kc, lane);
                acc[g] = __builtin_amdgcn_wmma_f32_16x16x32_bf16(
                    false, a, false, b, (short)0, acc[g], false, false);
            }
        }
    }
    const int col = jt * 16 + n;
    #pragma unroll
    for (int r = 0; r < 8; ++r) {
        int brow = mt * 16 + r + rsel;
        size_t gx = ((size_t)brow * TT + t) * G4;
        float iv = sigmoidf_(acc[0][r] + (float)gates_x[gx + 0 * HH + col]);
        float fv = sigmoidf_(acc[1][r] + (float)gates_x[gx + 1 * HH + col]);
        float gv = __sinf   (acc[2][r] + (float)gates_x[gx + 2 * HH + col]);
        float ov = sigmoidf_(acc[3][r] + (float)gates_x[gx + 3 * HH + col]);
        size_t ci = (size_t)brow * HH + col;
        float cnew = fv * c_state[ci] + iv * gv;
        c_state[ci] = cnew;
        float hnew = ov * __sinf(cnew);
        h_all[((size_t)brow * TT + t) * HH + col] = (__bf16)hnew;
    }
}

// ---------------- output projection: h_all[BT,512] @ Wo[512,32] + bo ----
__global__ void __launch_bounds__(64)
out_proj(const __bf16* __restrict__ h_all, const __bf16* __restrict__ Wop,
         const float* __restrict__ bo, float* __restrict__ out) {
    const int tid  = threadIdx.x;
    const int lane = tid & 31;
    const int w    = tid >> 5;            // N-tile 0 or 1
    const int mt   = blockIdx.x;          // 0..4095
    const int n    = lane & 15;
    const int rsel = (lane >> 4) << 3;

    const __bf16* hbase = h_all + (size_t)mt * 16 * HH;  // h_all is [BT][512]
    v8f acc = {};
    #pragma unroll 4
    for (int kc = 0; kc < 16; ++kc) {
        v16bf a = load_afrag(hbase + kc * 32, HH, lane);
        v16bf b = load_bfrag(Wop, w * 16 + kc, lane);
        acc = __builtin_amdgcn_wmma_f32_16x16x32_bf16(
            false, a, false, b, (short)0, acc, false, false);
    }
    float bias = bo[w * 16 + n];
    #pragma unroll
    for (int r = 0; r < 8; ++r)
        out[(size_t)(mt * 16 + r + rsel) * NS + w * 16 + n] = acc[r] + bias;
}

// ---------------- host launcher ----------------
extern "C" void kernel_launch(void* const* d_in, const int* in_sizes, int n_in,
                              void* d_out, int out_size, void* d_ws, size_t ws_size,
                              hipStream_t stream) {
    (void)in_sizes; (void)n_in; (void)out_size; (void)ws_size;
    const float* x      = (const float*)d_in[0];
    const float* W1     = (const float*)d_in[1];
    const float* b1     = (const float*)d_in[2];
    const float* W_ih   = (const float*)d_in[3];
    const float* W_hh   = (const float*)d_in[4];
    const float* b_lstm = (const float*)d_in[5];
    const float* Wo     = (const float*)d_in[6];
    const float* bo     = (const float*)d_in[7];
    float* out = (float*)d_out;

    // workspace layout (bytes, all 256-aligned)
    constexpr size_t SZ_W1P  = (size_t)II * HH * 2;        // 128 KB
    constexpr size_t SZ_WIHP = (size_t)HH * G4 * 2;        // 2 MB
    constexpr size_t SZ_WHHP = SZ_WIHP;                    // 2 MB
    constexpr size_t SZ_WOP  = (size_t)HH * NS * 2;        // 32 KB
    constexpr size_t SZ_C    = (size_t)BB * HH * 4;        // 128 KB
    constexpr size_t SZ_GX   = (size_t)BT * G4 * 2;        // 256 MB
    constexpr size_t OFF_W1P  = 0;
    constexpr size_t OFF_WIHP = OFF_W1P + SZ_W1P;
    constexpr size_t OFF_WHHP = OFF_WIHP + SZ_WIHP;
    constexpr size_t OFF_WOP  = OFF_WHHP + SZ_WHHP;
    constexpr size_t OFF_C    = OFF_WOP + SZ_WOP;
    constexpr size_t OFF_GX   = OFF_C + SZ_C;
    constexpr size_t OFF_HALL = OFF_GX + SZ_GX;            // + 64 MB

    char* ws = (char*)d_ws;
    __bf16* W1p   = (__bf16*)(ws + OFF_W1P);
    __bf16* Wihp  = (__bf16*)(ws + OFF_WIHP);
    __bf16* Whhp  = (__bf16*)(ws + OFF_WHHP);
    __bf16* Wop   = (__bf16*)(ws + OFF_WOP);
    float*  cst   = (float*)(ws + OFF_C);
    __bf16* gatesx= (__bf16*)(ws + OFF_GX);
    __bf16* hall  = (__bf16*)(ws + OFF_HALL);

    // 1) pack weights into bf16 B-fragment order
    pack_b_kernel<<<256, 256, 0, stream>>>(W1, W1p, II, HH);
    pack_b_kernel<<<2048, 256, 0, stream>>>(W_ih, Wihp, HH, G4);
    pack_b_kernel<<<2048, 256, 0, stream>>>(W_hh, Whhp, HH, G4);
    pack_b_kernel<<<64, 256, 0, stream>>>(Wo, Wop, HH, NS);

    // 2) zero cell state
    zero_f32_kernel<<<(BB * HH + 255) / 256, 256, 0, stream>>>(cst, BB * HH);

    // 3) fused fc1 + input-gate precompute (the big parallel GEMM)
    fused_fc1_gatesx<<<BT / 16, 128, 0, stream>>>(x, b1, b_lstm, W1p, Wihp, gatesx);

    // 4) sequential recurrence: 1024 steps, each h @ W_hh + gates_x + cell update
    for (int t = 0; t < TT; ++t)
        lstm_step<<<128, 32, 0, stream>>>(gatesx, Whhp, hall, cst, t);

    // 5) output projection
    out_proj<<<BT / 16, 64, 0, stream>>>(hall, Wop, bo, out);
}
